// QuantizedLinear_63513976373289
// MI455X (gfx1250) — matrix operand order repaired and checked by reading, not compiled, not run on hardware
//
#include <hip/hip_runtime.h>
#include <hip/hip_bf16.h>

#define B_   4
#define S_   2048
#define IN_  4096
#define OUT_ 16384
#define M_   (B_ * S_)

#define TM 128
#define TN 256

#define TK2  64
#define LDA2 72
#define LDB2 72
#define NK2  (IN_ / TK2)

#define TK1  32
#define LDA1 40
#define LDB1 40
#define NK1  (IN_ / TK1)

typedef __bf16 v16bf __attribute__((ext_vector_type(16)));
typedef float  v8f   __attribute__((ext_vector_type(8)));
typedef unsigned int v4u __attribute__((ext_vector_type(4)));
typedef unsigned int v8u __attribute__((ext_vector_type(8)));

// ---- CDNA5 async global->LDS path (probe via __has_builtin) ---------------
#if defined(__has_builtin)
#if __has_builtin(__builtin_amdgcn_global_load_async_to_lds_b128)
#define HAS_ASYNC_LDS 1
#endif
#endif
#ifndef HAS_ASYNC_LDS
#define HAS_ASYNC_LDS 0
#endif

#if HAS_ASYNC_LDS
// Prototype (from hipcc diagnostic): param0 = addrspace(1) v4i32*, then
// addrspace(3) dst, imm offset, imm cpol.
typedef int v4i_vs __attribute__((vector_size(16)));
typedef __attribute__((address_space(1))) v4i_vs* gv4_ptr;
typedef __attribute__((address_space(3))) v4i_vs* lv4_ptr;

__device__ __forceinline__ void async_b128(const void* g, void* l) {
    __builtin_amdgcn_global_load_async_to_lds_b128((gv4_ptr)g, (lv4_ptr)l, 0, 0);
}
__device__ __forceinline__ void wait_async0() {
#if __has_builtin(__builtin_amdgcn_s_wait_asynccnt)
    __builtin_amdgcn_s_wait_asynccnt(0);
#else
    asm volatile("s_wait_asynccnt 0x0" ::: "memory");
#endif
}
#endif

__device__ __forceinline__ unsigned int pack2(float lo, float hi) {
    __bf16 a = (__bf16)lo;
    __bf16 b = (__bf16)hi;
    unsigned int ua = __builtin_bit_cast(unsigned short, a);
    unsigned int ub = __builtin_bit_cast(unsigned short, b);
    return ua | (ub << 16);
}
__device__ __forceinline__ v16bf make_frag(v4u lo, v4u hi) {
    v8u t = __builtin_shufflevector(lo, hi, 0, 1, 2, 3, 4, 5, 6, 7);
    return __builtin_bit_cast(v16bf, t);
}

// Prep A: x -> bf16 + rowsum (single pass over x)
__global__ __launch_bounds__(256) void xprep_kernel(const float* __restrict__ x,
                                                    unsigned short* __restrict__ xb,
                                                    float* __restrict__ rs) {
    __shared__ float red[256];
    const int row = blockIdx.x;
    const float4* x4 = (const float4*)(x + (size_t)row * IN_);
    uint4* xb4 = (uint4*)(xb + (size_t)row * IN_);
    float s = 0.0f;
    #pragma unroll
    for (int it = 0; it < IN_ / 8 / 256; ++it) {
        const int i = threadIdx.x + 256 * it;
        float4 a = x4[2 * i];
        float4 b = x4[2 * i + 1];
        s += (a.x + a.y) + (a.z + a.w) + (b.x + b.y) + (b.z + b.w);
        uint4 o;
        o.x = pack2(a.x, a.y);
        o.y = pack2(a.z, a.w);
        o.z = pack2(b.x, b.y);
        o.w = pack2(b.z, b.w);
        xb4[i] = o;
    }
    red[threadIdx.x] = s;
    __syncthreads();
    for (int off = 128; off > 0; off >>= 1) {
        if ((int)threadIdx.x < off) red[threadIdx.x] += red[threadIdx.x + off];
        __syncthreads();
    }
    if (threadIdx.x == 0) rs[row] = red[0];
}

// Rowsum only (fallback path)
__global__ __launch_bounds__(256) void rowsum_kernel(const float* __restrict__ x,
                                                     float* __restrict__ rs) {
    __shared__ float red[256];
    const int row = blockIdx.x;
    const float4* x4 = (const float4*)(x + (size_t)row * IN_);
    float s = 0.0f;
    #pragma unroll
    for (int i = 0; i < IN_ / 4 / 256; ++i) {
        float4 f = x4[threadIdx.x + 256 * i];
        s += f.x + f.y + f.z + f.w;
    }
    red[threadIdx.x] = s;
    __syncthreads();
    for (int off = 128; off > 0; off >>= 1) {
        if ((int)threadIdx.x < off) red[threadIdx.x] += red[threadIdx.x + off];
        __syncthreads();
    }
    if (threadIdx.x == 0) rs[row] = red[0];
}

// Prep B: weight int32 -> bf16 (exact for 0..255)
__global__ __launch_bounds__(256) void wprep_kernel(const int* __restrict__ wq,
                                                    unsigned short* __restrict__ wb) {
    const size_t idx = ((size_t)blockIdx.x * 256 + threadIdx.x) * 8;
    int4 q0 = *(const int4*)(wq + idx);
    int4 q1 = *(const int4*)(wq + idx + 4);
    uint4 o;
    o.x = pack2((float)q0.x, (float)q0.y);
    o.y = pack2((float)q0.z, (float)q0.w);
    o.z = pack2((float)q1.x, (float)q1.y);
    o.w = pack2((float)q1.z, (float)q1.w);
    *(uint4*)(wb + idx) = o;
}

// ---------------------------------------------------------------------------
// Main GEMM: bf16 WMMA, block tile 128x256, wave tile 64x64, K-step 64.
// Async variant: double-buffered LDS fed by GLOBAL_LOAD_ASYNC_TO_LDS_B128,
// one barrier per K-step. Fallback: register-staged copy pipeline.
// ---------------------------------------------------------------------------
__global__ __launch_bounds__(256) void gemm_bf16_kernel(
    const unsigned short* __restrict__ xb, const unsigned short* __restrict__ wb,
    const float* __restrict__ scale, const int* __restrict__ zp,
    const float* __restrict__ bias, const float* __restrict__ rowsum,
    float* __restrict__ out) {

    const int tid   = threadIdx.x;
    const int lane  = tid & 31;
    const int wave  = tid >> 5;
    const int waveM = wave & 1;
    const int waveN = wave >> 1;
    const int mBase = blockIdx.y * TM;
    const int nBase = blockIdx.x * TN;

    const int aRow = tid >> 1;
    const int aK   = (tid & 1) * 32;
    const int bRow = tid;

    v8f acc[4][4];
    #pragma unroll
    for (int i = 0; i < 4; ++i)
        #pragma unroll
        for (int j = 0; j < 4; ++j)
            acc[i][j] = (v8f){0.f, 0.f, 0.f, 0.f, 0.f, 0.f, 0.f, 0.f};

    const int aHalf = lane >> 4;
    const int aLane = lane & 15;

#if HAS_ASYNC_LDS
    // ---------------- async double-buffered pipeline ----------------
    __shared__ __align__(16) unsigned short As[2][TM * LDA2];  // 2 x 18 KB
    __shared__ __align__(16) unsigned short Bs[2][TN * LDB2];  // 2 x 36 KB

    const unsigned short* gA = xb + (size_t)(mBase + aRow) * IN_ + aK;
    const unsigned short* gB = wb + (size_t)(nBase + bRow) * IN_;

    auto issue = [&](int kt, int buf) {
        const unsigned short* ga = gA + kt * TK2;
        unsigned short* la = &As[buf][aRow * LDA2 + aK];
        #pragma unroll
        for (int i = 0; i < 4; ++i) async_b128(ga + 8 * i, la + 8 * i);
        const unsigned short* gb = gB + kt * TK2;
        unsigned short* lb = &Bs[buf][bRow * LDB2];
        #pragma unroll
        for (int i = 0; i < 8; ++i) async_b128(gb + 8 * i, lb + 8 * i);
    };

    issue(0, 0);
    wait_async0();
    __syncthreads();
    issue(1, 1);

    for (int kt = 0; kt < NK2; ++kt) {
        const int buf = kt & 1;
        const unsigned short* Ab = As[buf];
        const unsigned short* Bb = Bs[buf];

        #pragma unroll
        for (int kk = 0; kk < 2; ++kk) {
            // Load all 8 fragments for this K=32 step, then 16 WMMAs.
            v16bf af[4], bfr[4];
            #pragma unroll
            for (int ti = 0; ti < 4; ++ti) {
                const unsigned short* base =
                    Ab + (waveM * 64 + ti * 16 + aLane) * LDA2 + kk * 32;
                v4u lo = *(const v4u*)(base + 16 * aHalf);
                v4u hi = *(const v4u*)(base + 16 * aHalf + 8);
                af[ti] = make_frag(lo, hi);
            }
            #pragma unroll
            for (int tj = 0; tj < 4; ++tj) {
                const unsigned short* base =
                    Bb + (waveN * 64 + tj * 16 + aLane) * LDB2 + kk * 32 +
                    16 * aHalf;
                v4u lo = *(const v4u*)(base);
                v4u hi = *(const v4u*)(base + 8);
                bfr[tj] = make_frag(lo, hi);
            }
            #pragma unroll
            for (int tj = 0; tj < 4; ++tj)
                #pragma unroll
                for (int ti = 0; ti < 4; ++ti)
                    acc[ti][tj] = __builtin_amdgcn_wmma_f32_16x16x32_bf16(
                        false, af[ti], false, bfr[tj], (short)0, acc[ti][tj],
                        false, false);
        }

        if (kt + 1 < NK2) {
            wait_async0();        // my kt+1 tile loads complete
            __syncthreads();      // everyone done reading buf & loading
            if (kt + 2 < NK2) issue(kt + 2, buf);
        }
    }
#else
    // ---------------- register-staged fallback pipeline ----------------
    __shared__ __align__(16) unsigned short As[TM * LDA2];
    __shared__ __align__(16) unsigned short Bs[TN * LDB2];

    const uint4* ag = (const uint4*)(xb + (size_t)(mBase + aRow) * IN_ + aK);
    const uint4* bg = (const uint4*)(wb + (size_t)(nBase + bRow) * IN_);

    uint4 aReg[4];
    uint4 bReg[8];
    #pragma unroll
    for (int i = 0; i < 4; ++i) aReg[i] = ag[i];
    #pragma unroll
    for (int i = 0; i < 8; ++i) bReg[i] = bg[i];

    for (int kt = 0; kt < NK2; ++kt) {
        {
            v4u* pa = (v4u*)(As + aRow * LDA2 + aK);
            #pragma unroll
            for (int i = 0; i < 4; ++i) pa[i] = __builtin_bit_cast(v4u, aReg[i]);
            v4u* pb = (v4u*)(Bs + bRow * LDB2);
            #pragma unroll
            for (int i = 0; i < 8; ++i) pb[i] = __builtin_bit_cast(v4u, bReg[i]);
        }
        __syncthreads();

        if (kt + 1 < NK2) {
            const uint4* an = ag + (kt + 1) * (TK2 / 8);
            const uint4* bn = bg + (kt + 1) * (TK2 / 8);
            #pragma unroll
            for (int i = 0; i < 4; ++i) aReg[i] = an[i];
            #pragma unroll
            for (int i = 0; i < 8; ++i) bReg[i] = bn[i];
        }
        if (kt + 2 < NK2) {  // near (WGP-scope) prefetch of tile kt+2
            __builtin_prefetch((const void*)(ag + (kt + 2) * (TK2 / 8)), 0, 3);
            __builtin_prefetch((const void*)(bg + (kt + 2) * (TK2 / 8)), 0, 3);
        }

        #pragma unroll
        for (int kk = 0; kk < 2; ++kk) {
            v16bf af[4];
            #pragma unroll
            for (int ti = 0; ti < 4; ++ti) {
                const unsigned short* base =
                    As + (waveM * 64 + ti * 16 + aLane) * LDA2 + kk * 32;
                v4u lo = *(const v4u*)(base + 16 * aHalf);
                v4u hi = *(const v4u*)(base + 16 * aHalf + 8);
                af[ti] = make_frag(lo, hi);
            }
            #pragma unroll
            for (int tj = 0; tj < 4; ++tj) {
                const unsigned short* base =
                    Bs + (waveN * 64 + tj * 16 + aLane) * LDB2 + kk * 32 +
                    16 * aHalf;
                v4u lo = *(const v4u*)(base);
                v4u hi = *(const v4u*)(base + 8);
                v16bf bf = make_frag(lo, hi);
                #pragma unroll
                for (int ti = 0; ti < 4; ++ti)
                    acc[ti][tj] = __builtin_amdgcn_wmma_f32_16x16x32_bf16(
                        false, af[ti], false, bf, (short)0, acc[ti][tj],
                        false, false);
            }
        }
        __syncthreads();
    }
#endif

    // Epilogue: out = scale*(acc - zp*rowsum) + bias
    const int half = lane >> 4;
    #pragma unroll
    for (int tj = 0; tj < 4; ++tj) {
        const int n  = nBase + waveN * 64 + tj * 16 + (lane & 15);
        const float sc  = scale[n];
        const float zpf = (float)zp[n];
        const float bi  = bias[n];
        #pragma unroll
        for (int ti = 0; ti < 4; ++ti) {
            const int m0 = mBase + waveM * 64 + ti * 16 + half * 8;
            #pragma unroll
            for (int r = 0; r < 8; ++r) {
                out[(size_t)(m0 + r) * OUT_ + n] =
                    sc * (acc[ti][tj][r] - zpf * rowsum[m0 + r]) + bi;
            }
        }
    }
}

// ---------------------------------------------------------------------------
// Fallback fused GEMM (used only if ws can't hold the bf16 copies)
// ---------------------------------------------------------------------------
__global__ __launch_bounds__(256) void qlinear_fused_kernel(
    const float* __restrict__ x, const int* __restrict__ wq,
    const float* __restrict__ scale, const int* __restrict__ zp,
    const float* __restrict__ bias, const float* __restrict__ rowsum,
    float* __restrict__ out) {

    __shared__ __align__(16) unsigned short As[TM * LDA1];
    __shared__ __align__(16) unsigned short Bs[TN * LDB1];

    const int tid   = threadIdx.x;
    const int lane  = tid & 31;
    const int wave  = tid >> 5;
    const int waveM = wave & 1;
    const int waveN = wave >> 1;
    const int mBase = blockIdx.y * TM;
    const int nBase = blockIdx.x * TN;

    const int aRow = tid >> 1;
    const int aK   = (tid & 1) * 16;
    const int bRow = tid;

    const float4* xg = (const float4*)(x + (size_t)(mBase + aRow) * IN_ + aK);
    const int4*   wg = (const int4*)(wq + (size_t)(nBase + bRow) * IN_);

    v8f acc[4][4];
    #pragma unroll
    for (int i = 0; i < 4; ++i)
        #pragma unroll
        for (int j = 0; j < 4; ++j)
            acc[i][j] = (v8f){0.f, 0.f, 0.f, 0.f, 0.f, 0.f, 0.f, 0.f};

    unsigned int aReg[8];
    unsigned int bReg[16];

    #pragma unroll
    for (int i = 0; i < 4; ++i) {
        float4 f = xg[i];
        aReg[2 * i]     = pack2(f.x, f.y);
        aReg[2 * i + 1] = pack2(f.z, f.w);
    }
    #pragma unroll
    for (int i = 0; i < 8; ++i) {
        int4 q = wg[i];
        bReg[2 * i]     = pack2((float)q.x, (float)q.y);
        bReg[2 * i + 1] = pack2((float)q.z, (float)q.w);
    }

    const int aHalf = lane >> 4;
    const int aLane = lane & 15;

    for (int kt = 0; kt < NK1; ++kt) {
        {
            v4u* pa = (v4u*)(As + aRow * LDA1 + aK);
            pa[0] = (v4u){aReg[0], aReg[1], aReg[2], aReg[3]};
            pa[1] = (v4u){aReg[4], aReg[5], aReg[6], aReg[7]};
            v4u* pb = (v4u*)(Bs + bRow * LDB1);
            #pragma unroll
            for (int i = 0; i < 4; ++i)
                pb[i] = (v4u){bReg[4 * i], bReg[4 * i + 1],
                              bReg[4 * i + 2], bReg[4 * i + 3]};
        }
        __syncthreads();

        if (kt + 1 < NK1) {
            const float4* xn = xg + (kt + 1) * (TK1 / 4);
            const int4*   wn = wg + (kt + 1) * (TK1 / 4);
            #pragma unroll
            for (int i = 0; i < 4; ++i) {
                float4 f = xn[i];
                aReg[2 * i]     = pack2(f.x, f.y);
                aReg[2 * i + 1] = pack2(f.z, f.w);
            }
            #pragma unroll
            for (int i = 0; i < 8; ++i) {
                int4 q = wn[i];
                bReg[2 * i]     = pack2((float)q.x, (float)q.y);
                bReg[2 * i + 1] = pack2((float)q.z, (float)q.w);
            }
        }

        v16bf af[4];
        #pragma unroll
        for (int ti = 0; ti < 4; ++ti) {
            const unsigned short* base =
                As + (waveM * 64 + ti * 16 + aLane) * LDA1;
            v4u lo = *(const v4u*)(base + 16 * aHalf);
            v4u hi = *(const v4u*)(base + 16 * aHalf + 8);
            af[ti] = make_frag(lo, hi);
        }
        #pragma unroll
        for (int tj = 0; tj < 4; ++tj) {
            const unsigned short* base =
                Bs + (waveN * 64 + tj * 16 + aLane) * LDB1 + 16 * aHalf;
            v4u lo = *(const v4u*)(base);
            v4u hi = *(const v4u*)(base + 8);
            v16bf bf = make_frag(lo, hi);
            #pragma unroll
            for (int ti = 0; ti < 4; ++ti)
                acc[ti][tj] = __builtin_amdgcn_wmma_f32_16x16x32_bf16(
                    false, af[ti], false, bf, (short)0, acc[ti][tj],
                    false, false);
        }
        __syncthreads();
    }

    const int half = lane >> 4;
    #pragma unroll
    for (int tj = 0; tj < 4; ++tj) {
        const int n  = nBase + waveN * 64 + tj * 16 + (lane & 15);
        const float sc  = scale[n];
        const float zpf = (float)zp[n];
        const float bi  = bias[n];
        #pragma unroll
        for (int ti = 0; ti < 4; ++ti) {
            const int m0 = mBase + waveM * 64 + ti * 16 + half * 8;
            #pragma unroll
            for (int r = 0; r < 8; ++r) {
                out[(size_t)(m0 + r) * OUT_ + n] =
                    sc * (acc[ti][tj][r] - zpf * rowsum[m0 + r]) + bi;
            }
        }
    }
}

extern "C" void kernel_launch(void* const* d_in, const int* in_sizes, int n_in,
                              void* d_out, int out_size, void* d_ws,
                              size_t ws_size, hipStream_t stream) {
    (void)in_sizes; (void)n_in; (void)out_size;
    const float* x     = (const float*)d_in[0];
    const int*   wq    = (const int*)d_in[1];
    const float* scale = (const float*)d_in[2];
    const int*   zp    = (const int*)d_in[3];
    const float* bias  = (const float*)d_in[4];
    float* out = (float*)d_out;

    const size_t needW = (size_t)OUT_ * IN_ * sizeof(unsigned short);
    const size_t needX = (size_t)M_ * IN_ * sizeof(unsigned short);
    const size_t needR = (size_t)M_ * sizeof(float);

    dim3 grid(OUT_ / TN, M_ / TM);

    if (ws_size >= needW + needX + needR) {
        unsigned short* wb = (unsigned short*)d_ws;
        unsigned short* xb = (unsigned short*)((char*)d_ws + needW);
        float* rowsum      = (float*)((char*)d_ws + needW + needX);
        wprep_kernel<<<((size_t)OUT_ * IN_) / 8 / 256, 256, 0, stream>>>(wq, wb);
        xprep_kernel<<<M_, 256, 0, stream>>>(x, xb, rowsum);
        gemm_bf16_kernel<<<grid, 256, 0, stream>>>(xb, wb, scale, zp, bias,
                                                   rowsum, out);
    } else {
        float* rowsum = (float*)d_ws;
        rowsum_kernel<<<M_, 256, 0, stream>>>(x, rowsum);
        qlinear_fused_kernel<<<grid, 256, 0, stream>>>(x, wq, scale, zp, bias,
                                                       rowsum, out);
    }
}
